// Yolov8NMS_7121055777082
// MI455X (gfx1250) — matrix operand order repaired and checked by reading, not compiled, not run on hardware
//
#include <hip/hip_runtime.h>

typedef __attribute__((ext_vector_type(16))) _Float16 v16h;
typedef __attribute__((ext_vector_type(8)))  float    v8f;

#define NA      8400
#define MAXD    300
#define MAXDP   304              // padded detections (19 tiles * 16)
#define NMASK   32
#define MH_     160
#define MW_     160
#define IMGSZ   640
#define NPIX    (MH_*MW_)        // 25600
#define OPIX    (IMGSZ*IMGSZ)    // 409600
#define MT_     19               // 304/16
#define NT_     (NPIX/16)        // 1600
#define ALPHA_  0.6f

// ---- workspace layout (float indices) ----
#define OFF_BOXES   0            // 8400*4
#define OFF_SCORE   33600        // 8400
#define OFF_BOXESK  42304        // 300*4
#define OFF_SCOREK  43504        // 300
#define OFF_VALIDK  43804        // 300
#define OFF_KEEP    44104        // 300
#define OFF_MK      44408        // 300*32
#define OFF_AW      54016        // f16: 19*512 halfs  (4864 floats)
#define OFF_PB      58880        // f16: 1600*512 halfs (409600 floats)
#define OFF_MLR     468480       // 304*25600 floats (padded low-res cropped masks)

// ---- output layout (float indices, tuple concat order) ----
#define OUT_RESULTS 0            // 300*6
#define OUT_KEEP    1800         // 300
#define OUT_MASKS   2100         // 300*640*640
#define OUT_PIXELS  122882100    // 1
#define OUT_IMG     122882101    // 3*640*640
#define OUT_CROP    124110901    // 4

__device__ __forceinline__ int imin_(int a, int b) { return a < b ? a : b; }
__device__ __forceinline__ int imax_(int a, int b) { return a > b ? a : b; }

// 16-bit A-matrix 16x32 layout (ISA 7.12.2): lane%16 = M,
// half j -> K = j%8 + (j/8)*16 + (lane/16)*8
__device__ __forceinline__ int kmapA(int lane, int j) {
    return (j & 7) + ((j >> 3) << 4) + ((lane >> 4) << 3);
}
// 16-bit B-matrix 32x16 layout: lane%16 = N, half j -> K = j%8 + (j/8)*16 + (lane/16)*8
__device__ __forceinline__ int kmapB(int lane, int j) {
    return (j & 7) + ((j >> 3) << 4) + ((lane >> 4) << 3);
}

// ---------------- K0: decode boxes + score, zero pixel counter ----------------
__global__ void k_decode(const float* __restrict__ pred,
                         const float* __restrict__ conf_thres,
                         float* __restrict__ ws, float* __restrict__ out) {
    int i = blockIdx.x * blockDim.x + threadIdx.x;
    if (i == 0) out[OUT_PIXELS] = 0.0f;
    if (i >= NA) return;
    float cx = pred[0 * NA + i], cy = pred[1 * NA + i];
    float w  = pred[2 * NA + i], h  = pred[3 * NA + i];
    ws[OFF_BOXES + i * 4 + 0] = cx - 0.5f * w;
    ws[OFF_BOXES + i * 4 + 1] = cy - 0.5f * h;
    ws[OFF_BOXES + i * 4 + 2] = cx + 0.5f * w;
    ws[OFF_BOXES + i * 4 + 3] = cy + 0.5f * h;
    float conf = pred[4 * NA + i];
    ws[OFF_SCORE + i] = (conf > conf_thres[0]) ? conf : -1.0f;
}

// ---------------- K1: stable rank -> top-300 gather ----------------
__global__ void k_topk(const float* __restrict__ pred,
                       const float* __restrict__ conf_thres,
                       float* __restrict__ ws) {
    __shared__ float sch[256];
    int i = blockIdx.x * blockDim.x + threadIdx.x;
    float si = (i < NA) ? ws[OFF_SCORE + i] : 0.0f;
    int rank = 0;
    for (int base = 0; base < NA; base += 256) {
        int j = base + threadIdx.x;
        sch[threadIdx.x] = (j < NA) ? ws[OFF_SCORE + j] : -2.0f;
        __syncthreads();
        int lim = imin_(256, NA - base);
        if (i < NA) {
            for (int t = 0; t < lim; ++t) {
                float sj = sch[t];
                int jj = base + t;
                if (sj > si || (sj == si && jj < i)) rank++;
            }
        }
        __syncthreads();
    }
    if (i < NA && rank < MAXD) {
        ws[OFF_BOXESK + rank * 4 + 0] = ws[OFF_BOXES + i * 4 + 0];
        ws[OFF_BOXESK + rank * 4 + 1] = ws[OFF_BOXES + i * 4 + 1];
        ws[OFF_BOXESK + rank * 4 + 2] = ws[OFF_BOXES + i * 4 + 2];
        ws[OFF_BOXESK + rank * 4 + 3] = ws[OFF_BOXES + i * 4 + 3];
        ws[OFF_SCOREK + rank] = si;
        ws[OFF_VALIDK + rank] = (si > conf_thres[0]) ? 1.0f : 0.0f;
        for (int c = 0; c < NMASK; ++c)
            ws[OFF_MK + rank * NMASK + c] = pred[(5 + c) * NA + i];
    }
}

// ---------------- K2a: build WMMA-swizzled f16 A tiles (coefs) ----------------
__global__ void k_build_a(float* __restrict__ ws) {
    int lane = threadIdx.x, mt = blockIdx.x;
    _Float16* Ah = (_Float16*)(ws + OFF_AW);
    int m = mt * 16 + (lane & 15);
    for (int j = 0; j < 16; ++j) {
        int k = kmapA(lane, j);
        float v = (m < MAXD) ? ws[OFF_MK + m * NMASK + k] : 0.0f;
        Ah[mt * 512 + lane * 16 + j] = (_Float16)v;
    }
}

// ---------------- K2b: build WMMA-swizzled f16 B tiles (proto) ----------------
__global__ void k_build_b(const float* __restrict__ proto, float* __restrict__ ws) {
    int lane = threadIdx.x, nt = blockIdx.x;
    _Float16* Bh = (_Float16*)(ws + OFF_PB);
    int n = nt * 16 + (lane & 15);
    for (int j = 0; j < 16; ++j) {
        int k = kmapB(lane, j);
        Bh[nt * 512 + lane * 16 + j] = (_Float16)proto[k * NPIX + n];
    }
}

// ---------------- K3: sequential NMS scan (single block) ----------------
__global__ void k_nms(const float* __restrict__ iou_thres,
                      float* __restrict__ ws, float* __restrict__ out) {
    __shared__ float bx0[MAXD], by0[MAXD], bx1[MAXD], by1[MAXD], ar[MAXD];
    __shared__ int skeep[MAXD];
    __shared__ int supp;
    int t = threadIdx.x;
    if (t < MAXD) {
        bx0[t] = ws[OFF_BOXESK + t * 4 + 0];
        by0[t] = ws[OFF_BOXESK + t * 4 + 1];
        bx1[t] = ws[OFF_BOXESK + t * 4 + 2];
        by1[t] = ws[OFF_BOXESK + t * 4 + 3];
        ar[t]  = (bx1[t] - bx0[t]) * (by1[t] - by0[t]);
        skeep[t] = 0;
    }
    __syncthreads();
    float thr = iou_thres[0];
    for (int i = 0; i < MAXD; ++i) {
        if (t == 0) supp = 0;
        __syncthreads();
        if (t < i && skeep[t]) {
            float iw = fmaxf(fminf(bx1[t], bx1[i]) - fmaxf(bx0[t], bx0[i]), 0.0f);
            float ih = fmaxf(fminf(by1[t], by1[i]) - fmaxf(by0[t], by0[i]), 0.0f);
            float inter = iw * ih;
            float iou = inter / (ar[t] + ar[i] - inter + 1e-7f);
            if (iou > thr) atomicOr(&supp, 1);
        }
        __syncthreads();
        if (t == i) skeep[i] = (ws[OFF_VALIDK + i] > 0.5f) && (supp == 0);
        __syncthreads();
    }
    if (t < MAXD) {
        float kf = (float)skeep[t];
        ws[OFF_KEEP + t] = kf;
        out[OUT_KEEP + t] = kf;
    }
}

// ---------------- K4: WMMA mask GEMM + sigmoid + crop ----------------
// Wave handles one M-tile x 4 N-tiles: A reused, 4 v_wmma per wave.
// Output rows padded to 304 so the epilogue stores are branch-free.
__global__ void __launch_bounds__(256) k_masks(float* __restrict__ ws) {
    int lane = threadIdx.x & 31;
    int wave = threadIdx.x >> 5;
    int wid  = blockIdx.x * 8 + wave;         // 0 .. 19*400-1
    int mt   = wid / 400;
    int ntg  = wid % 400;                     // group of 4 N-tiles
    const _Float16* Ah = (const _Float16*)(ws + OFF_AW);
    const _Float16* Bh = (const _Float16*)(ws + OFF_PB);

    v16h a = *(const v16h*)(Ah + mt * 512 + lane * 16);

    // hoist crop boxes for the 8 rows this lane produces
    int mbase = mt * 16 + ((lane >> 4) << 3);
    float db0[8], db1[8], db2[8], db3[8];
#pragma unroll
    for (int r = 0; r < 8; ++r) {
        int m = imin_(mbase + r, MAXD - 1);
        db0[r] = ws[OFF_BOXESK + m * 4 + 0] * 0.25f;   // MW/IMG
        db1[r] = ws[OFF_BOXESK + m * 4 + 1] * 0.25f;
        db2[r] = ws[OFF_BOXESK + m * 4 + 2] * 0.25f;
        db3[r] = ws[OFF_BOXESK + m * 4 + 3] * 0.25f;
    }

#pragma unroll
    for (int q = 0; q < 4; ++q) {
        int nt = ntg * 4 + q;
        v16h b = *(const v16h*)(Bh + nt * 512 + lane * 16);
        v8f c = {};
        c = __builtin_amdgcn_wmma_f32_16x16x32_f16(
                /*neg_a=*/false, a, /*neg_b=*/false, b,
                /*c_mod=*/(short)0, c, /*reuse_a=*/false, /*reuse_b=*/false);

        int n  = nt * 16 + (lane & 15);
        float fr = (float)(n % MW_);   // x
        float fc = (float)(n / MW_);   // y
#pragma unroll
        for (int r = 0; r < 8; ++r) {
            int m = mbase + r;                         // 0..303, always in padded range
            bool crop = (fr >= db0[r]) && (fr < db2[r]) &&
                        (fc >= db1[r]) && (fc < db3[r]);
            // fast sigmoid: v_exp_f32 + v_rcp_f32 (feeds a 0.5 threshold; f16 inputs)
            float sig = __builtin_amdgcn_rcpf(1.0f + __expf(-c[r]));
            ws[OFF_MLR + (size_t)m * NPIX + n] = crop ? sig : 0.0f;
        }
    }
}

// ---------------- K5: bilinear upsample + binarize + blend + count ----------------
__global__ void __launch_bounds__(256) k_upsample(const float* __restrict__ img,
                                                  const float* __restrict__ ws,
                                                  float* __restrict__ out) {
    __shared__ float skeep[MAXD];
    __shared__ float red[256];
    for (int t = threadIdx.x; t < MAXD; t += blockDim.x) skeep[t] = ws[OFF_KEEP + t];
    __syncthreads();

    int p  = blockIdx.x * blockDim.x + threadIdx.x;     // < OPIX exactly
    int px = p % IMGSZ, py = p / IMGSZ;
    float sx = ((float)px + 0.5f) * 0.25f - 0.5f;
    float sy = ((float)py + 0.5f) * 0.25f - 0.5f;
    float fx = floorf(sx), fy = floorf(sy);
    float wx = sx - fx, wy = sy - fy;
    int x0 = imin_(imax_((int)fx, 0), MW_ - 1);
    int x1 = imin_(imax_((int)fx + 1, 0), MW_ - 1);
    int y0 = imin_(imax_((int)fy, 0), MH_ - 1);
    int y1 = imin_(imax_((int)fy + 1, 0), MH_ - 1);
    float w00 = (1.0f - wx) * (1.0f - wy), w10 = wx * (1.0f - wy);
    float w01 = (1.0f - wx) * wy,          w11 = wx * wy;
    const float* mlr = ws + OFF_MLR;
    int o00 = y0 * MW_ + x0, o10 = y0 * MW_ + x1;
    int o01 = y1 * MW_ + x0, o11 = y1 * MW_ + x1;

    float stacked = 0.0f;
    for (int det = 0; det < MAXD; ++det) {
        const float* md = mlr + (size_t)det * NPIX;
        float v = w00 * md[o00] + w10 * md[o10] + w01 * md[o01] + w11 * md[o11];
        float bin = ((v > 0.5f) && (skeep[det] > 0.5f)) ? 1.0f : 0.0f;
        out[OUT_MASKS + (size_t)det * OPIX + p] = bin;
        stacked += bin;
    }

    float s = stacked;
    float i0 = img[0 * OPIX + p], i1 = img[1 * OPIX + p], i2 = img[2 * OPIX + p];
    out[OUT_IMG + 0 * OPIX + p] = (1.0f - s) * i0 + s * (ALPHA_ * i0);
    out[OUT_IMG + 1 * OPIX + p] = (1.0f - s) * i1 + s * (ALPHA_ * i1);
    out[OUT_IMG + 2 * OPIX + p] = (1.0f - s) * i2 + s * (ALPHA_ * i2 + (1.0f - ALPHA_));

    red[threadIdx.x] = (s > 0.0f) ? 1.0f : 0.0f;
    __syncthreads();
    for (int stp = 128; stp > 0; stp >>= 1) {
        if (threadIdx.x < stp) red[threadIdx.x] += red[threadIdx.x + stp];
        __syncthreads();
    }
    if (threadIdx.x == 0) atomicAdd(&out[OUT_PIXELS], red[0]);
}

// ---------------- K6: results / crop_coords ----------------
__global__ void k_finalize(const float* __restrict__ ws, float* __restrict__ out) {
    int t = threadIdx.x;
    if (t < MAXD) {
        float b0 = fminf(fmaxf(ws[OFF_BOXESK + t * 4 + 0], 0.0f), (float)IMGSZ);
        float b1 = fminf(fmaxf(ws[OFF_BOXESK + t * 4 + 1], 0.0f), (float)IMGSZ);
        float b2 = fminf(fmaxf(ws[OFF_BOXESK + t * 4 + 2], 0.0f), (float)IMGSZ);
        float b3 = fminf(fmaxf(ws[OFF_BOXESK + t * 4 + 3], 0.0f), (float)IMGSZ);
        float kf = ws[OFF_KEEP + t];
        out[OUT_RESULTS + t * 6 + 0] = b0 * kf;
        out[OUT_RESULTS + t * 6 + 1] = b1 * kf;
        out[OUT_RESULTS + t * 6 + 2] = b2 * kf;
        out[OUT_RESULTS + t * 6 + 3] = b3 * kf;
        out[OUT_RESULTS + t * 6 + 4] = ws[OFF_SCOREK + t] * kf;
        out[OUT_RESULTS + t * 6 + 5] = 0.0f;
    }
    if (t == 0) {
        float best = -1.0f; int bi = 0; bool any = false;
        for (int i = 0; i < MAXD; ++i) {
            float kf = ws[OFF_KEEP + i];
            float b0 = fminf(fmaxf(ws[OFF_BOXESK + i * 4 + 0], 0.0f), (float)IMGSZ);
            float b1 = fminf(fmaxf(ws[OFF_BOXESK + i * 4 + 1], 0.0f), (float)IMGSZ);
            float b2 = fminf(fmaxf(ws[OFF_BOXESK + i * 4 + 2], 0.0f), (float)IMGSZ);
            float b3 = fminf(fmaxf(ws[OFF_BOXESK + i * 4 + 3], 0.0f), (float)IMGSZ);
            float area = (kf > 0.5f) ? (b2 - b0) * (b3 - b1) : 0.0f;
            if (kf > 0.5f) any = true;
            if (area > best) { best = area; bi = i; }
        }
        float mb0 = 0.f, mb1 = 0.f, mb2 = 0.f, mb3 = 0.f;
        if (any) {
            mb0 = fminf(fmaxf(ws[OFF_BOXESK + bi * 4 + 0], 0.0f), (float)IMGSZ);
            mb1 = fminf(fmaxf(ws[OFF_BOXESK + bi * 4 + 1], 0.0f), (float)IMGSZ);
            mb2 = fminf(fmaxf(ws[OFF_BOXESK + bi * 4 + 2], 0.0f), (float)IMGSZ);
            mb3 = fminf(fmaxf(ws[OFF_BOXESK + bi * 4 + 3], 0.0f), (float)IMGSZ);
        }
        int cy1 = imax_((int)mb1 - 15, 0);
        int cy2 = imin_((int)mb3 + 15, IMGSZ);
        int cx1 = imax_((int)mb0 - 15, 0);
        int cx2 = imin_((int)mb2 + 15, IMGSZ);
        out[OUT_CROP + 0] = (float)cy1;
        out[OUT_CROP + 1] = (float)cy2;
        out[OUT_CROP + 2] = (float)cx1;
        out[OUT_CROP + 3] = (float)cx2;
    }
}

extern "C" void kernel_launch(void* const* d_in, const int* in_sizes, int n_in,
                              void* d_out, int out_size, void* d_ws, size_t ws_size,
                              hipStream_t stream) {
    (void)in_sizes; (void)n_in; (void)out_size; (void)ws_size;
    const float* pred       = (const float*)d_in[0];
    const float* proto      = (const float*)d_in[1];
    const float* img        = (const float*)d_in[2];
    const float* conf_thres = (const float*)d_in[3];
    const float* iou_thres  = (const float*)d_in[4];
    float* out = (float*)d_out;
    float* ws  = (float*)d_ws;

    k_decode  <<<(NA + 255) / 256, 256, 0, stream>>>(pred, conf_thres, ws, out);
    k_topk    <<<(NA + 255) / 256, 256, 0, stream>>>(pred, conf_thres, ws);
    k_build_a <<<MT_, 32, 0, stream>>>(ws);
    k_build_b <<<NT_, 32, 0, stream>>>(proto, ws);
    k_nms     <<<1, 320, 0, stream>>>(iou_thres, ws, out);
    k_masks   <<<(MT_ * 400) / 8, 256, 0, stream>>>(ws);          // 950 blocks, 4 WMMA/wave
    k_upsample<<<OPIX / 256, 256, 0, stream>>>(img, ws, out);
    k_finalize<<<1, 320, 0, stream>>>(ws, out);
}
